// DeepseekV3MoE_29506425323545
// MI455X (gfx1250) — matrix operand order repaired and checked by reading, not compiled, not run on hardware
//
#include <hip/hip_runtime.h>
#include <hip/hip_bf16.h>

// ---------------- problem constants ----------------
static constexpr int T    = 2048;   // tokens (B*S)
static constexpr int H    = 2048;   // hidden
static constexpr int IMOE = 1408;   // routed intermediate
static constexpr int E    = 8;      // experts
static constexpr int ISH  = 2816;   // shared intermediate

// ---------------- vector types ----------------
typedef __bf16 v16bf __attribute__((ext_vector_type(16)));
typedef __bf16 v8bf  __attribute__((ext_vector_type(8)));
typedef __bf16 v4bf  __attribute__((ext_vector_type(4)));
typedef float  v8f   __attribute__((ext_vector_type(8)));
typedef float  v4f   __attribute__((ext_vector_type(4)));

#define LDPX 56  // bf16 elems/row: 112B (16B-aligned, bank-spread)
#define LDPW 36  // f32  elems/row: 144B (16B-aligned, bank-spread)

__device__ __forceinline__ v16bf cat16(v8bf lo, v8bf hi) {
  return __builtin_shufflevector(lo, hi, 0,1,2,3,4,5,6,7,8,9,10,11,12,13,14,15);
}

// CDNA5 async copy: global -> LDS, 16B per lane, tracked by ASYNCcnt
__device__ __forceinline__ void async_b128(unsigned lds, unsigned goff, const void* base) {
  asm volatile("global_load_async_to_lds_b128 %0, %1, %2"
               :: "v"(lds), "v"(goff), "s"(base) : "memory");
}
__device__ __forceinline__ unsigned lds_off(const void* p) {
  return (unsigned)(unsigned long long)p;  // low 32 bits of generic addr = LDS offset
}

// ---------------- x fp32 -> bf16 ----------------
__global__ void k_cvt_x(const float* __restrict__ x, __bf16* __restrict__ xb) {
  int i = (blockIdx.x * blockDim.x + threadIdx.x) * 4;
  v4f v = *(const v4f*)(x + i);
  v4bf o;
#pragma unroll
  for (int j = 0; j < 4; ++j) o[j] = (__bf16)v[j];
  *(v4bf*)(xb + i) = o;
}

// ---------------- router: one wave per token ----------------
__global__ void k_router(const float* __restrict__ x, const float* __restrict__ rw,
                         const float* __restrict__ bias, float* __restrict__ cw) {
  const int wave = threadIdx.x >> 5;
  const int lane = threadIdx.x & 31;
  const int t = blockIdx.x * 8 + wave;
  const float* xr = x + (size_t)t * H;
  float acc[E];
#pragma unroll
  for (int e = 0; e < E; ++e) acc[e] = 0.f;
  for (int h = lane; h < H; h += 32) {
    float xv = xr[h];
#pragma unroll
    for (int e = 0; e < E; ++e) acc[e] += xv * rw[e * H + h];
  }
#pragma unroll
  for (int e = 0; e < E; ++e)
    for (int off = 16; off; off >>= 1) acc[e] += __shfl_xor(acc[e], off, 32);
  if (lane == 0) {
    float scores[E], sfc[E];
#pragma unroll
    for (int e = 0; e < E; ++e) {
      scores[e] = 1.0f / (1.0f + expf(-acc[e]));
      sfc[e] = scores[e] + bias[e];
    }
    float gs[4];
#pragma unroll
    for (int g = 0; g < 4; ++g) gs[g] = sfc[2 * g] + sfc[2 * g + 1];
    int g1 = 0;
    for (int g = 1; g < 4; ++g) if (gs[g] > gs[g1]) g1 = g;
    int g2 = -1;
    for (int g = 0; g < 4; ++g) if (g != g1 && (g2 < 0 || gs[g] > gs[g2])) g2 = g;
    float masked[E];
#pragma unroll
    for (int e = 0; e < E; ++e) {
      int g = e >> 1;
      masked[e] = (g == g1 || g == g2) ? sfc[e] : 0.0f;
    }
    int i1 = 0;
    for (int e = 1; e < E; ++e) if (masked[e] > masked[i1]) i1 = e;
    int i2 = -1;
    for (int e = 0; e < E; ++e) if (e != i1 && (i2 < 0 || masked[e] > masked[i2])) i2 = e;
    float w1 = scores[i1], w2 = scores[i2];
    float inv = 2.5f / (w1 + w2 + 1e-20f);
    float o[E];
#pragma unroll
    for (int e = 0; e < E; ++e) o[e] = 0.f;
    o[i1] = w1 * inv;
    o[i2] = w2 * inv;
#pragma unroll
    for (int e = 0; e < E; ++e) cw[t * E + e] = o[e];
  }
}

// ---------------- deterministic per-expert token lists ----------------
// 1 block, 8 waves; wave e compacts tokens with cw[t,e]>0 in token order.
__global__ void k_build_lists(const float* __restrict__ cw, int* __restrict__ tok,
                              int* __restrict__ cnt, int* __restrict__ cntp) {
  const int e = threadIdx.x >> 5;
  const int lane = threadIdx.x & 31;
  int n = 0;
  for (int base = 0; base < T; base += 32) {
    int t = base + lane;
    bool sel = cw[t * E + e] > 0.0f;
    unsigned m = (unsigned)__ballot(sel);
    int pre = __popc(m & ((1u << lane) - 1u));
    if (sel) tok[e * T + n + pre] = t;
    n += __popc(m);
  }
  int np = (n + 127) & ~127;
  for (int i = n + lane; i < np; i += 32) tok[e * T + i] = 0;  // pad with token 0
  if (lane == 0) { cnt[e] = n; cntp[e] = np; }
}

// ---------------- fused gate/up GEMM + SiLU*up (gathered rows) ----------------
// act[slot, n] = silu(x[tok[slot]]·gw[n]) * (x[tok[slot]]·uw[n])
__global__ __launch_bounds__(256) void k_gateup(
    const __bf16* __restrict__ xb, const float* __restrict__ gw,
    const float* __restrict__ uw, __bf16* __restrict__ act, int NI, int K,
    const int* __restrict__ tok, const int* __restrict__ cntp) {
  __shared__ __bf16 Xs[2][128 * LDPX];
  __shared__ float  Gs[2][128 * LDPW];
  __shared__ float  Us[2][128 * LDPW];
  const int m0 = blockIdx.x * 128;
  if (cntp && m0 >= cntp[0]) return;
  const int n0 = blockIdx.y * 128;
  const int tid = threadIdx.x;
  const int lane = tid & 31, wave = tid >> 5;
  const int wm = wave & 1, wn = wave >> 1;
  const int hl = lane >> 4, l15 = lane & 15;

  // per-thread async chunk plumbing (row assignments invariant over K-loop)
  const int xr = tid >> 2, xcc = (tid & 3) * 8;                 // X: 2 chunks
  const int tA = tok ? tok[m0 + xr]      : (m0 + xr);
  const int tB = tok ? tok[m0 + xr + 64] : (m0 + xr + 64);
  const unsigned xgA = ((unsigned)tA * (unsigned)K + (unsigned)xcc) * 2u;
  const unsigned xgB = ((unsigned)tB * (unsigned)K + (unsigned)xcc) * 2u;
  const unsigned xlA = lds_off(&Xs[0][xr * LDPX + xcc]);
  const unsigned xlB = lds_off(&Xs[0][(xr + 64) * LDPX + xcc]);
  const unsigned xst = (unsigned)sizeof(Xs[0]);
  const int wr = tid >> 3, wcc = (tid & 7) * 4;                 // W: 4 rows x (G,U)
  unsigned wg[4], gl[4], ul[4];
#pragma unroll
  for (int i = 0; i < 4; ++i) {
    int r = wr + i * 32;
    wg[i] = ((unsigned)(n0 + r) * (unsigned)K + (unsigned)wcc) * 4u;
    gl[i] = lds_off(&Gs[0][r * LDPW + wcc]);
    ul[i] = lds_off(&Us[0][r * LDPW + wcc]);
  }
  const unsigned wst = (unsigned)sizeof(Gs[0]);

  auto issue = [&](int st, int kb) {   // 10 async b128 per thread per stage
    const unsigned xo = (unsigned)kb * 2u, wo = (unsigned)kb * 4u;
    const unsigned xs = st ? xst : 0u, wso = st ? wst : 0u;
    async_b128(xlA + xs, xgA + xo, xb);
    async_b128(xlB + xs, xgB + xo, xb);
#pragma unroll
    for (int i = 0; i < 4; ++i) {
      async_b128(gl[i] + wso, wg[i] + wo, gw);
      async_b128(ul[i] + wso, wg[i] + wo, uw);
    }
  };

  const v8f vz = {0.f,0.f,0.f,0.f,0.f,0.f,0.f,0.f};
  v8f accG[4][2], accU[4][2];
#pragma unroll
  for (int i = 0; i < 4; ++i)
#pragma unroll
    for (int j = 0; j < 2; ++j) { accG[i][j] = vz; accU[i][j] = vz; }

  const int NK = K / 32;
  issue(0, 0);
  for (int j = 0; j < NK; ++j) {
    const int st = j & 1;
    if (j + 1 < NK) {
      issue(st ^ 1, (j + 1) * 32);
      asm volatile("s_wait_asynccnt 10" ::: "memory");  // stage j done, j+1 in flight
    } else {
      asm volatile("s_wait_asynccnt 0" ::: "memory");
    }
    __syncthreads();

    v16bf bG[2], bU[2];
#pragma unroll
    for (int ns = 0; ns < 2; ++ns) {
      const float* gp = &Gs[st][(wn * 32 + ns * 16 + l15) * LDPW + hl * 16];
      const float* up = &Us[st][(wn * 32 + ns * 16 + l15) * LDPW + hl * 16];
#pragma unroll
      for (int q = 0; q < 4; ++q) {
        v4f g4 = *(const v4f*)(gp + q * 4);
        v4f u4 = *(const v4f*)(up + q * 4);
#pragma unroll
        for (int c = 0; c < 4; ++c) {
          bG[ns][q * 4 + c] = (__bf16)g4[c];
          bU[ns][q * 4 + c] = (__bf16)u4[c];
        }
      }
    }
#pragma unroll
    for (int ms = 0; ms < 4; ++ms) {
      const __bf16* ap = &Xs[st][(wm * 64 + ms * 16 + l15) * LDPX];
      v16bf a = cat16(*(const v8bf*)(ap + hl * 16), *(const v8bf*)(ap + hl * 16 + 8));
#pragma unroll
      for (int ns = 0; ns < 2; ++ns) {
        accG[ms][ns] = __builtin_amdgcn_wmma_f32_16x16x32_bf16(
            false, a, false, bG[ns], (short)0, accG[ms][ns], false, false);
        accU[ms][ns] = __builtin_amdgcn_wmma_f32_16x16x32_bf16(
            false, a, false, bU[ns], (short)0, accU[ms][ns], false, false);
      }
    }
    __syncthreads();
  }

#pragma unroll
  for (int ms = 0; ms < 4; ++ms)
#pragma unroll
    for (int ns = 0; ns < 2; ++ns)
#pragma unroll
      for (int r = 0; r < 8; ++r) {
        float g = accG[ms][ns][r];
        float u = accU[ms][ns][r];
        float s = g / (1.0f + __expf(-g));
        int slot = m0 + wm * 64 + ms * 16 + hl * 8 + r;
        int col = n0 + wn * 32 + ns * 16 + l15;
        act[(size_t)slot * NI + col] = (__bf16)(s * u);
      }
}

// ---------------- down GEMM, scatter-accumulate into out ----------------
__global__ __launch_bounds__(256) void k_down(
    const __bf16* __restrict__ act, const float* __restrict__ dw,
    const float* __restrict__ cwcol, float* __restrict__ out, int K, int init,
    const int* __restrict__ tok, const int* __restrict__ cnt,
    const int* __restrict__ cntp) {
  __shared__ __bf16 As[2][128 * LDPX];
  __shared__ float  Ds[2][128 * LDPW];
  const int m0 = blockIdx.x * 128;
  if (cntp && m0 >= cntp[0]) return;
  const int n0 = blockIdx.y * 128;
  const int tid = threadIdx.x;
  const int lane = tid & 31, wave = tid >> 5;
  const int wm = wave & 1, wn = wave >> 1;
  const int hl = lane >> 4, l15 = lane & 15;

  const int xr = tid >> 2, xcc = (tid & 3) * 8;
  const unsigned agA = ((unsigned)(m0 + xr) * (unsigned)K + (unsigned)xcc) * 2u;
  const unsigned agB = ((unsigned)(m0 + xr + 64) * (unsigned)K + (unsigned)xcc) * 2u;
  const unsigned alA = lds_off(&As[0][xr * LDPX + xcc]);
  const unsigned alB = lds_off(&As[0][(xr + 64) * LDPX + xcc]);
  const unsigned ast = (unsigned)sizeof(As[0]);
  const int wr = tid >> 3, wcc = (tid & 7) * 4;
  unsigned wg[4], dl[4];
#pragma unroll
  for (int i = 0; i < 4; ++i) {
    int r = wr + i * 32;
    wg[i] = ((unsigned)(n0 + r) * (unsigned)K + (unsigned)wcc) * 4u;
    dl[i] = lds_off(&Ds[0][r * LDPW + wcc]);
  }
  const unsigned wst = (unsigned)sizeof(Ds[0]);

  auto issue = [&](int st, int kb) {   // 6 async b128 per thread per stage
    const unsigned ao = (unsigned)kb * 2u, wo = (unsigned)kb * 4u;
    const unsigned as = st ? ast : 0u, wso = st ? wst : 0u;
    async_b128(alA + as, agA + ao, act);
    async_b128(alB + as, agB + ao, act);
#pragma unroll
    for (int i = 0; i < 4; ++i) async_b128(dl[i] + wso, wg[i] + wo, dw);
  };

  const v8f vz = {0.f,0.f,0.f,0.f,0.f,0.f,0.f,0.f};
  v8f acc[4][2];
#pragma unroll
  for (int i = 0; i < 4; ++i)
#pragma unroll
    for (int j = 0; j < 2; ++j) acc[i][j] = vz;

  const int NK = K / 32;
  issue(0, 0);
  for (int j = 0; j < NK; ++j) {
    const int st = j & 1;
    if (j + 1 < NK) {
      issue(st ^ 1, (j + 1) * 32);
      asm volatile("s_wait_asynccnt 6" ::: "memory");
    } else {
      asm volatile("s_wait_asynccnt 0" ::: "memory");
    }
    __syncthreads();

    v16bf bD[2];
#pragma unroll
    for (int ns = 0; ns < 2; ++ns) {
      const float* dp = &Ds[st][(wn * 32 + ns * 16 + l15) * LDPW + hl * 16];
#pragma unroll
      for (int q = 0; q < 4; ++q) {
        v4f d4 = *(const v4f*)(dp + q * 4);
#pragma unroll
        for (int c = 0; c < 4; ++c) bD[ns][q * 4 + c] = (__bf16)d4[c];
      }
    }
#pragma unroll
    for (int ms = 0; ms < 4; ++ms) {
      const __bf16* ap = &As[st][(wm * 64 + ms * 16 + l15) * LDPX];
      v16bf a = cat16(*(const v8bf*)(ap + hl * 16), *(const v8bf*)(ap + hl * 16 + 8));
#pragma unroll
      for (int ns = 0; ns < 2; ++ns)
        acc[ms][ns] = __builtin_amdgcn_wmma_f32_16x16x32_bf16(
            false, a, false, bD[ns], (short)0, acc[ms][ns], false, false);
    }
    __syncthreads();
  }

  const int cv = cnt ? cnt[0] : (1 << 30);
#pragma unroll
  for (int ms = 0; ms < 4; ++ms)
#pragma unroll
    for (int ns = 0; ns < 2; ++ns)
#pragma unroll
      for (int r = 0; r < 8; ++r) {
        int slot = m0 + wm * 64 + ms * 16 + hl * 8 + r;
        int col = n0 + wn * 32 + ns * 16 + l15;
        if (slot < cv) {
          int t = tok ? tok[slot] : slot;
          float w = cwcol ? cwcol[(size_t)t * E] : 1.0f;
          float v = w * acc[ms][ns][r];
          size_t oi = (size_t)t * H + col;
          if (init) out[oi] = v; else out[oi] += v;
        }
      }
}

// ---------------- launch ----------------
extern "C" void kernel_launch(void* const* d_in, const int* in_sizes, int n_in,
                              void* d_out, int out_size, void* d_ws, size_t ws_size,
                              hipStream_t stream) {
  const float* x    = (const float*)d_in[0];
  const float* rw   = (const float*)d_in[1];
  const float* bias = (const float*)d_in[2];
  const float* gwp  = (const float*)d_in[3];
  const float* uwp  = (const float*)d_in[4];
  const float* dwp  = (const float*)d_in[5];
  const float* sgw  = (const float*)d_in[6];
  const float* suw  = (const float*)d_in[7];
  const float* sdw  = (const float*)d_in[8];
  float* out = (float*)d_out;

  char* ws = (char*)d_ws;
  size_t off = 0;
  __bf16* xb = (__bf16*)(ws + off); off += (size_t)T * H * 2;       // 8 MB
  float* cw  = (float*)(ws + off);  off += (size_t)T * E * 4;       // 64 KB
  int* tok   = (int*)(ws + off);    off += (size_t)E * T * 4;       // 64 KB
  int* cnt   = (int*)(ws + off);    off += 256;
  int* cntp  = (int*)(ws + off);    off += 256;
  __bf16* act = (__bf16*)(ws + off);                                // up to T*ISH*2

  k_cvt_x<<<(T * H) / (256 * 4), 256, 0, stream>>>(x, xb);
  k_router<<<T / 8, 256, 0, stream>>>(x, rw, bias, cw);
  k_build_lists<<<1, 256, 0, stream>>>(cw, tok, cnt, cntp);

  // shared expert first: dense, initializes all of out
  k_gateup<<<dim3(T / 128, ISH / 128), 256, 0, stream>>>(
      xb, sgw, suw, act, ISH, H, nullptr, nullptr);
  k_down<<<dim3(T / 128, H / 128), 256, 0, stream>>>(
      act, sdw, nullptr, out, ISH, /*init=*/1, nullptr, nullptr, nullptr);

  // routed experts: gathered token lists, accumulate in fixed order
  for (int e = 0; e < E; ++e) {
    k_gateup<<<dim3(T / 128, IMOE / 128), 256, 0, stream>>>(
        xb, gwp + (size_t)e * IMOE * H, uwp + (size_t)e * IMOE * H,
        act, IMOE, H, tok + (size_t)e * T, cntp + e);
    k_down<<<dim3(T / 128, H / 128), 256, 0, stream>>>(
        act, dwp + (size_t)e * H * IMOE, cw + e, out, IMOE, /*init=*/0,
        tok + (size_t)e * T, cnt + e, cntp + e);
  }
}
